// MultiHeadAttention_41223096107029
// MI455X (gfx1250) — matrix operand order repaired and checked by reading, not compiled, not run on hardware
//
#include <hip/hip_runtime.h>
#include <hip/hip_bf16.h>

// ---------------------------------------------------------------------------
// Causal MHA forward for B=2,T=2048,E=768,H=12,HD=64 on gfx1250 (MI455X).
// ~71 GFLOP vs ~35 MB traffic -> matrix-engine bound; everything is L2
// resident (192 MB). All GEMMs + attention run on v_wmma_f32_16x16x32_f16
// (f16 in, f32 accum); softmax is f32 online (flash) per 16-query tile.
//
// Round-3 changes (from disasm feedback):
//  * Attention load issue order fixed: V(current) issued BEFORE K(next).
//    LOADcnt decrements in issue order, so P@V now waits with loadcnt<=8
//    (V only) while the next block's K tiles stay in flight across the loop
//    edge, hidden behind softmax + P@V + loop-top work.
//  * GEMM K-loop: clamp/select removed; proper peeled software pipeline
//    (loads at kk feed WMMAs of kk-32; last WMMA group peeled).
// ---------------------------------------------------------------------------

typedef __attribute__((ext_vector_type(4)))  _Float16 v4h;
typedef __attribute__((ext_vector_type(8)))  _Float16 v8h;
typedef __attribute__((ext_vector_type(16))) _Float16 v16h;
typedef __attribute__((ext_vector_type(8)))  float    v8f;

#define B_  2
#define T_  2048
#define E_  768
#define H_  12
#define HD_ 64
#define MT_ (B_ * T_)   // 4096 rows of X

// ---- WMMA wrapper: D = A(16x32 f16) * B(32x16 f16) + C(16x16 f32) ----------
__device__ __forceinline__ v8f wmma32(v16h a, v16h b, v8f c) {
  return __builtin_amdgcn_wmma_f32_16x16x32_f16(
      /*neg_a=*/false, a, /*neg_b=*/false, b,
      /*c_mod=*/(short)0, c, /*reuse_a=*/false, /*reuse_b=*/false);
}

// ---- A-operand load (16x32, row-major source, ld in elements) --------------
// CDNA5 ISA "16-bit A-Matrix 16x32": lane L holds row m=L%16;
// element j -> K = 16*(j>>3) + 8*(L>>4) + (j&7)  => two contiguous v8h chunks.
__device__ __forceinline__ v16h loadA16(const _Float16* __restrict__ base,
                                        int row0, int ld, int k0, int lane) {
  const int m = lane & 15, half = lane >> 4;
  const _Float16* p = base + (size_t)(row0 + m) * ld + k0 + 8 * half;
  v8h lo = *(const v8h*)p;
  v8h hi = *(const v8h*)(p + 16);
  return __builtin_shufflevector(lo, hi, 0, 1, 2, 3, 4, 5, 6, 7,
                                 8, 9, 10, 11, 12, 13, 14, 15);
}

// ---- B-operand load (32x16). B[k,n] taken from row (col0+n) of a row-major
// matrix with leading dim ld: lane L holds column n=L%16,
// K = 16*(j>>3) + 8*(L>>4) + (j&7) (same as A) => two contiguous v8h chunks.
__device__ __forceinline__ v16h loadB16(const _Float16* __restrict__ base,
                                        int col0, int ld, int k0, int lane) {
  const int n = lane & 15, half = lane >> 4;
  const _Float16* p = base + (size_t)(col0 + n) * ld + k0 + 8 * half;
  v8h lo = *(const v8h*)p;
  v8h hi = *(const v8h*)(p + 16);
  return __builtin_shufflevector(lo, hi, 0, 1, 2, 3, 4, 5, 6, 7,
                                 8, 9, 10, 11, 12, 13, 14, 15);
}

// ---- f32 -> f16 conversion, 4 elements/thread ------------------------------
__global__ __launch_bounds__(256) void cvt_f32_f16(const float* __restrict__ in,
                                                   _Float16* __restrict__ out,
                                                   int n4) {
  int i = blockIdx.x * 256 + threadIdx.x;
  if (i < n4) {
    float4 v = ((const float4*)in)[i];
    v4h o = {(_Float16)v.x, (_Float16)v.y, (_Float16)v.z, (_Float16)v.w};
    ((v4h*)out)[i] = o;
  }
}

// ---- GEMM: out = A(MxK) @ W^T, W is (NxK) row-major. -----------------------
// MODE 0: f16 out, row-major (MxN).  MODE 1: f16 out scattered into
// (B,H,HD,T) (for V).  MODE 2: f32 out row-major (final result).
// Block = 256 thr = 8 waves in 4(M) x 2(N); wave tile 32x32; block 128x64.
// Peeled software pipeline: loads at kk feed the WMMAs of step kk-32.
template <int MODE>
__global__ __launch_bounds__(256) void gemm_wmma(
    const _Float16* __restrict__ A, const _Float16* __restrict__ W,
    void* __restrict__ out, int M, int N, int K) {
  const int lane = threadIdx.x & 31;
  const int wave = threadIdx.x >> 5;
  const int row0 = blockIdx.x * 128 + (wave & 3) * 32;
  const int col0 = blockIdx.y * 64 + (wave >> 2) * 32;

  v8f acc[2][2];
#pragma unroll
  for (int i = 0; i < 2; ++i)
#pragma unroll
    for (int j = 0; j < 2; ++j)
      acc[i][j] = v8f{0.f, 0.f, 0.f, 0.f, 0.f, 0.f, 0.f, 0.f};

  v16h a0 = loadA16(A, row0, K, 0, lane);
  v16h a1 = loadA16(A, row0 + 16, K, 0, lane);
  v16h b0 = loadB16(W, col0, K, 0, lane);
  v16h b1 = loadB16(W, col0 + 16, K, 0, lane);

  for (int kk = 32; kk < K; kk += 32) {
    // Speculative prefetch one K-step past the loads below
    // (global_prefetch_b8; OOB translation failures are silently dropped).
    __builtin_prefetch(A + (size_t)(row0 + (lane & 15)) * K + kk + 32, 0, 3);
    __builtin_prefetch(W + (size_t)(col0 + (lane & 15)) * K + kk + 32, 0, 3);
    v16h na0 = loadA16(A, row0, K, kk, lane);
    v16h na1 = loadA16(A, row0 + 16, K, kk, lane);
    v16h nb0 = loadB16(W, col0, K, kk, lane);
    v16h nb1 = loadB16(W, col0 + 16, K, kk, lane);
    acc[0][0] = wmma32(a0, b0, acc[0][0]);
    acc[0][1] = wmma32(a0, b1, acc[0][1]);
    acc[1][0] = wmma32(a1, b0, acc[1][0]);
    acc[1][1] = wmma32(a1, b1, acc[1][1]);
    a0 = na0; a1 = na1; b0 = nb0; b1 = nb1;
  }
  // Peeled final K-step.
  acc[0][0] = wmma32(a0, b0, acc[0][0]);
  acc[0][1] = wmma32(a0, b1, acc[0][1]);
  acc[1][0] = wmma32(a1, b0, acc[1][0]);
  acc[1][1] = wmma32(a1, b1, acc[1][1]);

  const int n = lane & 15, half = lane >> 4;
#pragma unroll
  for (int i = 0; i < 2; ++i)
#pragma unroll
    for (int j = 0; j < 2; ++j)
#pragma unroll
      for (int r = 0; r < 8; ++r) {
        const int mr = row0 + i * 16 + r + 8 * half;  // global row (b*T + t)
        const int nc = col0 + j * 16 + n;             // output feature
        const float v = acc[i][j][r];
        if (MODE == 2) {
          ((float*)out)[(size_t)mr * N + nc] = v;
        } else if (MODE == 0) {
          ((_Float16*)out)[(size_t)mr * N + nc] = (_Float16)v;
        } else {
          // Scatter V into (B,H,HD,T). Per-batch flat idx f = t*E + nc; the
          // reference reshape gives h = f >> 17 (T*HD = 2^17), then
          // t_chunk = (f & 131071) >> 6, d = f & 63.
          const int b = mr >> 11;          // T_ = 2048
          const int t = mr & 2047;
          const int f = t * E_ + nc;
          const int hh = f >> 17;
          const int idx = f & 131071;
          const int tc = idx >> 6;
          const int d = idx & 63;
          ((_Float16*)out)[((size_t)(b * H_ + hh) * HD_ + d) * T_ + tc] =
              (_Float16)v;
        }
      }
}

// ---- Flash attention over one (b, h, 16-query tile); 1 wave / block --------
__global__ __launch_bounds__(32) void attn_wmma(
    const _Float16* __restrict__ Qh, const _Float16* __restrict__ Kh,
    const _Float16* __restrict__ Vt, _Float16* __restrict__ At) {
  const int lane = threadIdx.x;
  const int n = lane & 15, half = lane >> 4;
  const int qbase = blockIdx.x * 16;
  const int h = blockIdx.y;
  const int b = blockIdx.z;

  const size_t chunk = (size_t)(b * H_ + h) * (T_ * HD_);
  const _Float16* Qc = Qh + chunk;  // (T, HD) row-major
  const _Float16* Kc = Kh + chunk;  // (T, HD) row-major
  const _Float16* Vc = Vt + chunk;  // (HD, T) row-major (pre-transposed)

  __shared__ __align__(32) _Float16 P[16 * 32];

  // Q operands (resident across the key loop): K-dim = HD = 64 -> 2 steps.
  const v16h aq0 = loadA16(Qc, qbase, HD_, 0, lane);
  const v16h aq1 = loadA16(Qc, qbase, HD_, 32, lane);

  v8f acc[4];
#pragma unroll
  for (int i = 0; i < 4; ++i)
    acc[i] = v8f{0.f, 0.f, 0.f, 0.f, 0.f, 0.f, 0.f, 0.f};

  float mrow[8], lrow[8];
#pragma unroll
  for (int r = 0; r < 8; ++r) { mrow[r] = -3.0e38f; lrow[r] = 0.f; }

  const float L2E = 1.44269504088896340736f;  // log2(e)
  const int kend = ((qbase + 15) / 32) * 32;  // last key-block base (incl.)

  // Prime the K-operand double buffer with key block 0.
  v16h bk0 = loadB16(Kc, 0, HD_, 0, lane);
  v16h bk1 = loadB16(Kc, 0, HD_, 32, lane);
  v16h bk2 = loadB16(Kc, 16, HD_, 0, lane);
  v16h bk3 = loadB16(Kc, 16, HD_, 32, lane);

  for (int kbase = 0; kbase <= kend; kbase += 32) {
    // ---- S = (Q @ K^T) * rsqrt(HD) from the pre-loaded K operands ---------
    v8f s0 = v8f{0.f, 0.f, 0.f, 0.f, 0.f, 0.f, 0.f, 0.f};
    v8f s1 = v8f{0.f, 0.f, 0.f, 0.f, 0.f, 0.f, 0.f, 0.f};
    s0 = wmma32(aq0, bk0, s0);
    s0 = wmma32(aq1, bk1, s0);
    s1 = wmma32(aq0, bk2, s1);
    s1 = wmma32(aq1, bk3, s1);

    // ---- issue THIS block's V loads FIRST, then NEXT block's K loads.
    // LOADcnt decrements in issue order: the P@V wait below only drains the
    // (older) V loads; the K loads stay in flight across the loop edge.
    v16h bv0 = loadB16(Vc, 0, T_, kbase, lane);
    v16h bv1 = loadB16(Vc, 16, T_, kbase, lane);
    v16h bv2 = loadB16(Vc, 32, T_, kbase, lane);
    v16h bv3 = loadB16(Vc, 48, T_, kbase, lane);
    // Unconditional next-block K load. On the last iteration this over-reads
    // at most ~4KB past the K chunk into the adjacent (mapped) Vt workspace
    // buffer; the values are never consumed.
    const int kn = kbase + 32;
    v16h nk0 = loadB16(Kc, kn, HD_, 0, lane);
    v16h nk1 = loadB16(Kc, kn, HD_, 32, lane);
    v16h nk2 = loadB16(Kc, kn + 16, HD_, 0, lane);
    v16h nk3 = loadB16(Kc, kn + 16, HD_, 32, lane);

#pragma unroll
    for (int r = 0; r < 8; ++r) { s0[r] *= 0.125f; s1[r] *= 0.125f; }

    // ---- causal mask (chunk-row causality, faithful to the reference) -----
    if (kbase + 31 > qbase) {
#pragma unroll
      for (int r = 0; r < 8; ++r) {
        const int q = qbase + r + 8 * half;
        if (kbase + n > q)      s0[r] = -3.0e30f;
        if (kbase + 16 + n > q) s1[r] = -3.0e30f;
      }
    }

    // ---- online softmax stats: butterfly over the 16-lane half-wave -------
    float rm[8];
#pragma unroll
    for (int r = 0; r < 8; ++r) rm[r] = fmaxf(s0[r], s1[r]);
#pragma unroll
    for (int off = 1; off <= 8; off <<= 1)
#pragma unroll
      for (int r = 0; r < 8; ++r)
        rm[r] = fmaxf(rm[r], __shfl_xor(rm[r], off, 32));

    float corr[8];
#pragma unroll
    for (int r = 0; r < 8; ++r) {
      const float mn = fmaxf(mrow[r], rm[r]);
      corr[r] = exp2f((mrow[r] - mn) * L2E);
      mrow[r] = mn;
    }

    float rs[8];
#pragma unroll
    for (int r = 0; r < 8; ++r) {
      s0[r] = exp2f((s0[r] - mrow[r]) * L2E);
      s1[r] = exp2f((s1[r] - mrow[r]) * L2E);
      rs[r] = s0[r] + s1[r];
    }
#pragma unroll
    for (int off = 1; off <= 8; off <<= 1)
#pragma unroll
      for (int r = 0; r < 8; ++r) rs[r] += __shfl_xor(rs[r], off, 32);
#pragma unroll
    for (int r = 0; r < 8; ++r) lrow[r] = lrow[r] * corr[r] + rs[r];

#pragma unroll
    for (int i = 0; i < 4; ++i)
#pragma unroll
      for (int r = 0; r < 8; ++r) acc[i][r] *= corr[r];

    // ---- P: C-layout -> A-layout through LDS ------------------------------
#pragma unroll
    for (int r = 0; r < 8; ++r) {
      const int row = r + 8 * half;
      P[row * 32 + n]      = (_Float16)s0[r];
      P[row * 32 + 16 + n] = (_Float16)s1[r];
    }
    asm volatile("s_wait_dscnt 0" ::: "memory");  // DS store->load, same wave
    v16h ap;
    {
      const _Float16* pr = &P[n * 32 + 8 * half];
      v8h lo = *(const v8h*)pr;
      v8h hi = *(const v8h*)(pr + 16);
      ap = __builtin_shufflevector(lo, hi, 0, 1, 2, 3, 4, 5, 6, 7,
                                   8, 9, 10, 11, 12, 13, 14, 15);
    }

    // ---- O += P @ V (waits only for the V loads; K stays in flight) -------
    acc[0] = wmma32(ap, bv0, acc[0]);
    acc[1] = wmma32(ap, bv1, acc[1]);
    acc[2] = wmma32(ap, bv2, acc[2]);
    acc[3] = wmma32(ap, bv3, acc[3]);

    bk0 = nk0; bk1 = nk1; bk2 = nk2; bk3 = nk3;
  }

  // ---- normalize and store into (B, T, H*HD) f16 --------------------------
#pragma unroll
  for (int r = 0; r < 8; ++r) {
    const float inv = 1.f / lrow[r];
    const int q = qbase + r + 8 * half;
    const size_t rowoff = (size_t)(b * T_ + q) * E_ + h * HD_;
#pragma unroll
    for (int nt = 0; nt < 4; ++nt)
      At[rowoff + nt * 16 + n] = (_Float16)(acc[nt][r] * inv);
  }
}

// ---------------------------------------------------------------------------
extern "C" void kernel_launch(void* const* d_in, const int* in_sizes, int n_in,
                              void* d_out, int out_size, void* d_ws,
                              size_t ws_size, hipStream_t stream) {
  const float* X  = (const float*)d_in[0];
  const float* Wq = (const float*)d_in[1];
  const float* Wk = (const float*)d_in[2];
  const float* Wv = (const float*)d_in[3];
  const float* Wp = (const float*)d_in[4];
  float* out = (float*)d_out;

  const size_t szX = (size_t)MT_ * E_;  // 3,145,728 halves
  const size_t szW = (size_t)E_ * E_;   //   589,824 halves

  _Float16* ws16 = (_Float16*)d_ws;     // total ~36.2 MB of f16 scratch
  _Float16* Xh  = ws16;
  _Float16* Wqh = Xh + szX;
  _Float16* Wkh = Wqh + szW;
  _Float16* Wvh = Wkh + szW;
  _Float16* Wph = Wvh + szW;
  _Float16* Qh  = Wph + szW;
  _Float16* Kh  = Qh + szX;
  _Float16* Vt  = Kh + szX;
  _Float16* At  = Vt + szX;

  // 1) f32 -> f16 conversions
  cvt_f32_f16<<<(int)(szX / 4 + 255) / 256, 256, 0, stream>>>(X, Xh, (int)(szX / 4));
  cvt_f32_f16<<<(int)(szW / 4 + 255) / 256, 256, 0, stream>>>(Wq, Wqh, (int)(szW / 4));
  cvt_f32_f16<<<(int)(szW / 4 + 255) / 256, 256, 0, stream>>>(Wk, Wkh, (int)(szW / 4));
  cvt_f32_f16<<<(int)(szW / 4 + 255) / 256, 256, 0, stream>>>(Wv, Wvh, (int)(szW / 4));
  cvt_f32_f16<<<(int)(szW / 4 + 255) / 256, 256, 0, stream>>>(Wp, Wph, (int)(szW / 4));

  // 2) projections
  dim3 gg(MT_ / 128, E_ / 64);
  gemm_wmma<0><<<gg, 256, 0, stream>>>(Xh, Wqh, (void*)Qh, MT_, E_, E_);
  gemm_wmma<0><<<gg, 256, 0, stream>>>(Xh, Wkh, (void*)Kh, MT_, E_, E_);
  gemm_wmma<1><<<gg, 256, 0, stream>>>(Xh, Wvh, (void*)Vt, MT_, E_, E_);

  // 3) flash attention: (q-tiles, heads, batch), 1 wave per block
  attn_wmma<<<dim3(T_ / 16, H_, B_), 32, 0, stream>>>(Qh, Kh, Vt, At);

  // 4) pooled output GEMM, f32 epilogue to d_out
  gemm_wmma<2><<<gg, 256, 0, stream>>>(At, Wph, (void*)out, MT_, E_, E_);
}
